// CausalSelfAttention_22943715295865
// MI455X (gfx1250) — matrix operand order repaired and checked by reading, not compiled, not run on hardware
//
#include <hip/hip_runtime.h>
#include <hip/hip_bf16.h>

// ---------------------------------------------------------------------------
// CausalSelfAttention forward for MI455X (gfx1250, wave32, WMMA bf16 + TDM).
// B=4, T=2048, C=1024, H=16, D=64.  All matmuls via v_wmma_f32_16x16x32_bf16;
// LDS tile staging via the Tensor Data Mover (tensor_load_to_lds).
// ---------------------------------------------------------------------------

typedef float        v8f   __attribute__((ext_vector_type(8)));
typedef __bf16       v16bf __attribute__((ext_vector_type(16)));
typedef unsigned int u32x4 __attribute__((ext_vector_type(4)));
typedef int          i32x4 __attribute__((ext_vector_type(4)));
typedef int          i32x8 __attribute__((ext_vector_type(8)));

#define WMMA_BF16(a, b, c) \
  __builtin_amdgcn_wmma_f32_16x16x32_bf16(false, (a), false, (b), (short)0, (c), false, false)

// ---------------------------------------------------------------------------
// TDM 2D tile load: global (row-major, bf16) -> LDS, with optional row padding.
// Descriptor bit layout per CDNA5 ISA §8.3 (group 0) / §8.4 (group 1).
//   pad_int_code: LDS pad interval = 2^(code+1) DWORDs;  pad_amt_code: pad =
//   (code+1) DWORDs.
// ---------------------------------------------------------------------------
__device__ __forceinline__ void tdm_load_2d_bf16(unsigned lds_addr,
                                                 const void* gptr,
                                                 unsigned rows, unsigned rowlen,
                                                 unsigned stride_elems,
                                                 unsigned pad_int_code,
                                                 unsigned pad_amt_code) {
  unsigned long long ga = (unsigned long long)(size_t)gptr;
  u32x4 g0;
  g0[0] = 1u;                                    // count=1, is_restore=0
  g0[1] = lds_addr;                              // LDS byte address
  g0[2] = (unsigned)(ga & 0xffffffffu);          // global_addr[31:0]
  g0[3] = (unsigned)((ga >> 32) & 0x01ffffffu)   // global_addr[56:32]
          | 0x80000000u;                         // type=2 ("image")
  i32x8 g1;
  unsigned w0 = (1u << 16)                       // data_size = 2 bytes
                | (1u << 20)                     // pad_enable
                | (pad_int_code << 22) | (pad_amt_code << 25);
  g1[0] = (int)w0;                               // workgroup_mask=0 (no cluster)
  g1[1] = (int)((rowlen & 0xffffu) << 16);       // tensor_dim0[15:0]
  g1[2] = (int)(((rowlen >> 16) & 0xffffu) | ((rows & 0xffffu) << 16));
  g1[3] = (int)(((rows >> 16) & 0xffffu) | ((rowlen & 0xffffu) << 16)); // tile_dim0
  g1[4] = (int)(rows & 0xffffu);                 // tile_dim1 (tile_dim2 = 0)
  g1[5] = (int)stride_elems;                     // tensor_dim0_stride[31:0]
  g1[6] = 0;
  g1[7] = 0;
  i32x4 z4 = {0, 0, 0, 0};
  i32x8 z8 = {0, 0, 0, 0, 0, 0, 0, 0};
  // amdgpu-toolchain (clang-23) 6-arg form: (g0, g1, g2, g3, g_extra, cpol)
  __builtin_amdgcn_tensor_load_to_lds(g0, g1, z4, z4, z8, 0);
}

// ---------------------------------------------------------------------------
// fp32 -> bf16 conversion (n must be a multiple of 4)
// ---------------------------------------------------------------------------
__global__ void f32_to_bf16_kernel(const float* __restrict__ in,
                                   __bf16* __restrict__ out, int n) {
  int i = (blockIdx.x * blockDim.x + threadIdx.x) * 4;
  if (i < n) {
    float4 v = *reinterpret_cast<const float4*>(in + i);
    __bf16 o4[4] = {(__bf16)v.x, (__bf16)v.y, (__bf16)v.z, (__bf16)v.w};
    *reinterpret_cast<uint2*>(out + i) = *reinterpret_cast<const uint2*>(o4);
  }
}

// ---------------------------------------------------------------------------
// GEMM: out[M x N] = A[M x K] * W[N x K]^T + bias[N]
// Block tile 128(M) x 128(N), K-step 32 through LDS (staged by TDM).
// 8 waves in a 4x2 grid; each wave owns 32x64 = 2x4 WMMA accumulators
// (8 WMMAs per K-step from 6 fragment loads).
// ---------------------------------------------------------------------------
template <bool OUTF32>
__global__ __launch_bounds__(256)
void gemm_bf16_wmma(const __bf16* __restrict__ A,
                    const __bf16* __restrict__ W,
                    const float* __restrict__ bias,
                    float* __restrict__ outF,
                    __bf16* __restrict__ outB,
                    int M, int N, int K) {
  constexpr int BM = 128, BN = 128, BK = 32, LD = BK + 8;  // pad -> stride 40
  __shared__ __align__(16) __bf16 As[BM][LD];
  __shared__ __align__(16) __bf16 Ws[BN][LD];

  const int tid  = threadIdx.x;
  const int lane = tid & 31;
  const int wave = tid >> 5;      // 0..7
  const int wm   = wave >> 1;     // 0..3 -> 32-row sub-tile
  const int wn   = wave & 1;      // 0..1 -> 64-col sub-tile
  const int lh   = lane >> 4;     // half: 0/1
  const int l16  = lane & 15;
  const int m0   = blockIdx.x * BM;
  const int n0   = blockIdx.y * BN;

  const unsigned ldsA = (unsigned)(size_t)&As[0][0];
  const unsigned ldsW = (unsigned)(size_t)&Ws[0][0];

  v8f acc[2][4] = {};

  for (int k0 = 0; k0 < K; k0 += BK) {
    __syncthreads();
    if (wave == 0) {
      // BK=32 bf16 rows = 16 DWORDs -> pad_interval code 3 (16 DW),
      // pad_amount code 3 (4 DW = 8 bf16) -> LDS row stride 40 bf16.
      tdm_load_2d_bf16(ldsA, A + (size_t)m0 * K + k0, BM, BK, (unsigned)K, 3, 3);
      tdm_load_2d_bf16(ldsW, W + (size_t)n0 * K + k0, BN, BK, (unsigned)K, 3, 3);
      __builtin_amdgcn_s_wait_tensorcnt(0);
    }
    __syncthreads();

    // A-fragment (16x32 bf16): lane l<16 row M=l holds K={0..7,16..23};
    // lanes 16..31 hold K={8..15,24..31}.
    v16bf afrag[2], bfrag[4];
#pragma unroll
    for (int mt = 0; mt < 2; ++mt) {
      const int row = 32 * wm + 16 * mt + l16;
#pragma unroll
      for (int j = 0; j < 8; ++j) {
        afrag[mt][j]     = As[row][8 * lh + j];
        afrag[mt][j + 8] = As[row][16 + 8 * lh + j];
      }
    }
    // B-fragment (32x16 bf16): lane&15 = column N, elems j -> K = 8*half + (j&7) + 16*(j>>3)
#pragma unroll
    for (int nt = 0; nt < 4; ++nt) {
      const int col = 64 * wn + 16 * nt + l16;
#pragma unroll
      for (int j = 0; j < 16; ++j) bfrag[nt][j] = Ws[col][8 * lh + (j & 7) + 16 * (j >> 3)];
    }
#pragma unroll
    for (int mt = 0; mt < 2; ++mt)
#pragma unroll
      for (int nt = 0; nt < 4; ++nt)
        acc[mt][nt] = WMMA_BF16(afrag[mt], bfrag[nt], acc[mt][nt]);
  }

  // Epilogue. C/D layout: vals[r] = (row r + 8*half, col lane&15).
#pragma unroll
  for (int mt = 0; mt < 2; ++mt)
#pragma unroll
    for (int nt = 0; nt < 4; ++nt) {
      const int col = n0 + 64 * wn + 16 * nt + l16;
      const float bv = bias ? bias[col] : 0.0f;
#pragma unroll
      for (int r = 0; r < 8; ++r) {
        const int row = m0 + 32 * wm + 16 * mt + 8 * lh + r;
        const float v = acc[mt][nt][r] + bv;
        if (OUTF32) outF[(size_t)row * N + col] = v;
        else        outB[(size_t)row * N + col] = (__bf16)v;
      }
    }
}

// ---------------------------------------------------------------------------
// Flash-attention forward, one wave per (b, h, 16-row q-tile).
// Online softmax over 32-key chunks; S and P*V via bf16 WMMA.
// K tile staged by TDM; V staged with a register-scatter transpose.
// ---------------------------------------------------------------------------
__global__ __launch_bounds__(32)
void attn_fwd_wmma(const __bf16* __restrict__ qkv,  // [B*T][3C] bf16
                   __bf16* __restrict__ out) {      // [B*T][C]  bf16
  constexpr int T = 2048, C = 1024, C3 = 3072, D = 64;
  const int qt = blockIdx.x & 127;   // T/16 = 128 q-tiles
  const int bh = blockIdx.x >> 7;
  const int h  = bh & 15;
  const int b  = bh >> 4;
  const int q0 = qt * 16;

  const int lane = threadIdx.x;
  const int lh   = lane >> 4;
  const int l16  = lane & 15;

  __shared__ __align__(16) __bf16 Klds[32][72];  // [key][d] (TDM-padded rows)
  __shared__ __align__(16) __bf16 Vt[64][40];    // [d][key], transposed
  __shared__ __align__(16) __bf16 Plds[16][40];  // [q][key]

  const __bf16* Qg = qkv + (size_t)(b * T + q0) * C3 + h * D;
  const __bf16* Kg = qkv + (size_t)(b * T) * C3 + C + h * D;
  const __bf16* Vg = qkv + (size_t)(b * T) * C3 + 2 * C + h * D;
  const unsigned ldsK = (unsigned)(size_t)&Klds[0][0];

  // Q fragments for the two 32-wide d-chunks, held for the whole tile.
  v16bf qfrag[2];
#pragma unroll
  for (int c = 0; c < 2; ++c) {
    const __bf16* qr = Qg + (size_t)l16 * C3 + 32 * c;
#pragma unroll
    for (int j = 0; j < 8; ++j) {
      qfrag[c][j]     = qr[8 * lh + j];
      qfrag[c][j + 8] = qr[16 + 8 * lh + j];
    }
  }

  v8f o[4] = {};            // 16 x 64 f32 output accumulator (4 d-tiles)
  float m_run[8], l_run[8]; // per-row running max / sum (rows r + 8*half)
#pragma unroll
  for (int r = 0; r < 8; ++r) { m_run[r] = -1e30f; l_run[r] = 0.0f; }

  const int nChunks = (q0 + 16 + 31) / 32;
  for (int ch = 0; ch < nChunks; ++ch) {
    const int j0 = ch * 32;
    __syncthreads();
    // K chunk 32x64 via TDM: 64 bf16 = 32 DWORDs -> pad_interval code 4
    // (32 DW), pad_amount code 3 (4 DW = 8 bf16) -> LDS row stride 72 bf16.
    tdm_load_2d_bf16(ldsK, Kg + (size_t)j0 * C3, 32, 64, C3, 4, 3);
    {   // stage V chunk transposed: vectorized load, scattered LDS store
      const uint4* g =
          reinterpret_cast<const uint4*>(Vg + (size_t)(j0 + lane) * C3);
      uint4 tmp[8];
#pragma unroll
      for (int i = 0; i < 8; ++i) tmp[i] = g[i];
      const __bf16* tb = reinterpret_cast<const __bf16*>(tmp);
#pragma unroll
      for (int d = 0; d < 64; ++d) Vt[d][lane] = tb[d];
    }
    __builtin_amdgcn_s_wait_tensorcnt(0);
    __syncthreads();

    // S(16x32) = Q(16x64) @ K^T(64x32): 2 key-tiles x 2 d-chunks = 4 WMMA
    v8f s[2] = {};
#pragma unroll
    for (int nt = 0; nt < 2; ++nt) {
      v16bf kf0, kf1;
#pragma unroll
      for (int j = 0; j < 16; ++j) {
        kf0[j] = Klds[16 * nt + l16][8 * lh + (j & 7) + 16 * (j >> 3)];
        kf1[j] = Klds[16 * nt + l16][32 + 8 * lh + (j & 7) + 16 * (j >> 3)];
      }
      s[nt] = WMMA_BF16(qfrag[0], kf0, s[nt]);
      s[nt] = WMMA_BF16(qfrag[1], kf1, s[nt]);
    }

    // scale, causal mask, online softmax (row = r + 8*half, col = lane&15)
    float p0[8], p1[8];
#pragma unroll
    for (int r = 0; r < 8; ++r) {
      const int qrow = q0 + 8 * lh + r;
      const int k0i = j0 + l16, k1i = j0 + 16 + l16;
      p0[r] = (k0i <= qrow) ? s[0][r] * 0.125f : -1e30f;
      p1[r] = (k1i <= qrow) ? s[1][r] * 0.125f : -1e30f;
    }
#pragma unroll
    for (int r = 0; r < 8; ++r) {
      float mx = fmaxf(p0[r], p1[r]);
#pragma unroll
      for (int off = 1; off < 16; off <<= 1)
        mx = fmaxf(mx, __shfl_xor(mx, off, 32));
      const float mnew  = fmaxf(m_run[r], mx);
      const float scale = __expf(m_run[r] - mnew);
      const float e0 = __expf(p0[r] - mnew);
      const float e1 = __expf(p1[r] - mnew);
      float rs = e0 + e1;
#pragma unroll
      for (int off = 1; off < 16; off <<= 1) rs += __shfl_xor(rs, off, 32);
      l_run[r] = l_run[r] * scale + rs;
      m_run[r] = mnew;
#pragma unroll
      for (int dt = 0; dt < 4; ++dt) o[dt][r] *= scale;
      p0[r] = e0; p1[r] = e1;
    }

    // P (C/D layout) -> LDS row-major -> A-fragment layout
    __syncthreads();
#pragma unroll
    for (int r = 0; r < 8; ++r) {
      Plds[8 * lh + r][l16]      = (__bf16)p0[r];
      Plds[8 * lh + r][16 + l16] = (__bf16)p1[r];
    }
    __syncthreads();
    v16bf pf;
#pragma unroll
    for (int j = 0; j < 8; ++j) {
      pf[j]     = Plds[l16][8 * lh + j];
      pf[j + 8] = Plds[l16][16 + 8 * lh + j];
    }
    // O(16x64) += P(16x32) @ V(32x64): 4 d-tiles
#pragma unroll
    for (int dt = 0; dt < 4; ++dt) {
      v16bf vf;
#pragma unroll
      for (int j = 0; j < 16; ++j) vf[j] = Vt[16 * dt + l16][8 * lh + (j & 7) + 16 * (j >> 3)];
      o[dt] = WMMA_BF16(pf, vf, o[dt]);
    }
  }

  // finalize: divide by row sums, write bf16 [B*T][C] head-interleaved
#pragma unroll
  for (int r = 0; r < 8; ++r) {
    const int qrow = q0 + 8 * lh + r;
    const float inv = 1.0f / l_run[r];
#pragma unroll
    for (int dt = 0; dt < 4; ++dt) {
      const int d = 16 * dt + l16;
      out[(size_t)(b * T + qrow) * C + h * D + d] = (__bf16)(o[dt][r] * inv);
    }
  }
}

// ---------------------------------------------------------------------------
// Launch
// ---------------------------------------------------------------------------
extern "C" void kernel_launch(void* const* d_in, const int* in_sizes, int n_in,
                              void* d_out, int out_size, void* d_ws,
                              size_t ws_size, hipStream_t stream) {
  constexpr int B = 4, T = 2048, C = 1024, C3 = 3072;
  constexpr int M = B * T;                       // 8192 rows
  constexpr size_t N_X   = (size_t)M * C;        // 8,388,608
  constexpr size_t N_WA  = (size_t)C3 * C;       // 3,145,728
  constexpr size_t N_WP  = (size_t)C * C;        // 1,048,576
  constexpr size_t N_QKV = (size_t)M * C3;       // 25,165,824
  constexpr size_t N_AO  = (size_t)M * C;        // 8,388,608

  const float* x      = (const float*)d_in[0];
  const float* w_attn = (const float*)d_in[1];
  const float* b_attn = (const float*)d_in[2];
  const float* w_proj = (const float*)d_in[3];
  const float* b_proj = (const float*)d_in[4];
  float* out = (float*)d_out;

  char* ws = (char*)d_ws;
  size_t off = 0;
  __bf16* x_bf   = (__bf16*)(ws + off); off += N_X   * 2;
  __bf16* wa_bf  = (__bf16*)(ws + off); off += N_WA  * 2;
  __bf16* wp_bf  = (__bf16*)(ws + off); off += N_WP  * 2;
  __bf16* qkv_bf = (__bf16*)(ws + off); off += N_QKV * 2;
  __bf16* ao_bf  = (__bf16*)(ws + off); off += N_AO  * 2;  // ~88 MB total

  const int cvt_blk = 256;
  f32_to_bf16_kernel<<<(int)((N_X / 4 + cvt_blk - 1) / cvt_blk), cvt_blk, 0,
                       stream>>>(x, x_bf, (int)N_X);
  f32_to_bf16_kernel<<<(int)((N_WA / 4 + cvt_blk - 1) / cvt_blk), cvt_blk, 0,
                       stream>>>(w_attn, wa_bf, (int)N_WA);
  f32_to_bf16_kernel<<<(int)((N_WP / 4 + cvt_blk - 1) / cvt_blk), cvt_blk, 0,
                       stream>>>(w_proj, wp_bf, (int)N_WP);

  // qkv = x @ w_attn^T + b_attn  (M=8192, N=3072, K=1024), bf16 out
  gemm_bf16_wmma<false><<<dim3(M / 128, C3 / 128), 256, 0, stream>>>(
      x_bf, wa_bf, b_attn, nullptr, qkv_bf, M, C3, C);

  // flash attention: one wave per (b, h, 16-row q-tile) -> 8192 blocks
  attn_fwd_wmma<<<B * 16 * (T / 16), 32, 0, stream>>>(qkv_bf, ao_bf);

  // y = attn_out @ w_proj^T + b_proj  (M=8192, N=1024, K=1024), f32 out
  gemm_bf16_wmma<true><<<dim3(M / 128, C / 128), 256, 0, stream>>>(
      ao_bf, wp_bf, b_proj, out, nullptr, M, C, C);
}
